// BA_86852828660053
// MI455X (gfx1250) — matrix operand order repaired and checked by reading, not compiled, not run on hardware
//
#include <hip/hip_runtime.h>

typedef __attribute__((ext_vector_type(16))) __bf16 v16bf;
typedef __attribute__((ext_vector_type(8)))  float  v8f;

#define WMMA_BF16(A, B, C) \
  __builtin_amdgcn_wmma_f32_16x16x32_bf16(false, (A), false, (B), (short)0, (C), false, false)

// 16-byte async global->LDS copy (CDNA5, tracked on ASYNCcnt).
__device__ __forceinline__ void async_copy16(const void* gptr, void* lptr) {
  unsigned int lds = (unsigned int)(unsigned long long)lptr;  // low 32 bits = LDS offset
  asm volatile("global_load_async_to_lds_b128 %0, %1, off"
               :: "v"(lds), "v"(gptr) : "memory");
}
__device__ __forceinline__ void wait_async0() {
  asm volatile("s_wait_asynccnt 0x0" ::: "memory");
}

// ---------------- Kernel 0: zero q/k sum accumulators ----------------
__global__ void BA_zero_kernel(float* __restrict__ p, int n) {
  int i = blockIdx.x * 256 + threadIdx.x;
  if (i < n) p[i] = 0.0f;
}

// ---------------- Kernel 1: windowed QKV projection (WMMA) ----------------
// grid = 64 windows * 8 token-blocks, block = 256 (8 waves). Each wave: 16 tokens x 192 outs.
__global__ __launch_bounds__(256)
void BA_qkv_kernel(const float* __restrict__ x, const float* __restrict__ W,
                   const float* __restrict__ bias,
                   __bf16* __restrict__ qb, __bf16* __restrict__ kb,
                   __bf16* __restrict__ vtb,
                   float* __restrict__ qsumG, float* __restrict__ ksumG) {
  __shared__ __bf16 Wl[192 * 72];     // W as bf16, padded row stride 72 (bank-conflict-free)
  __shared__ float qsL[64], ksL[64];

  const int tid = threadIdx.x;
  for (int idx = tid; idx < 192 * 64; idx += 256) {
    const int n = idx >> 6, k = idx & 63;
    Wl[n * 72 + k] = (__bf16)W[idx];
  }
  if (tid < 64) { qsL[tid] = 0.0f; ksL[tid] = 0.0f; }
  __syncthreads();

  const int bid  = blockIdx.x;
  const int win  = bid >> 3;
  const int mblk = bid & 7;
  const int wh0 = (win >> 3) << 5, ww0 = (win & 7) << 5;
  const int wv = tid >> 5, lane = tid & 31, tloc = lane & 15;
  const int t0 = mblk * 128 + wv * 16;

  // ---- A operand: 16 tokens x 64 channels gathered from CHW x, cvt to bf16 ----
  const int gt = t0 + tloc;                        // token in window (A rows: lanes 0-15 == 16-31)
  const int h = wh0 + (gt >> 5), w = ww0 + (gt & 31);
  const float* xp = x + h * 256 + w;
  const int khalfA = (lane >> 4) << 3;             // 0 or 8
  v16bf A[2];
#pragma unroll
  for (int kk = 0; kk < 2; ++kk) {
#pragma unroll
    for (int e = 0; e < 16; ++e) {
      int k = kk * 32 + khalfA + e + ((e >= 8) ? 8 : 0);
      A[kk][e] = (__bf16)xp[k << 16];              // stride 65536 floats per channel
    }
  }

  const int khB = (lane >> 4) << 3;                // 0 or 8
  const int mh  = (lane >> 4) << 3;                // D-layout row half
  const size_t wbase = (size_t)win << 16;          // 65536 elements per window tensor

  for (int nt = 0; nt < 12; ++nt) {
    const int n0 = nt << 4;
    const int n  = n0 + tloc;
    const float bval = bias[n];
    v8f acc;
#pragma unroll
    for (int e = 0; e < 8; ++e) acc[e] = bval;

#pragma unroll
    for (int kk = 0; kk < 2; ++kk) {
      const __bf16* bp = &Wl[n * 72 + kk * 32 + khB];  // k runs 8h..8h+7 and 16+8h.., 16B aligned
      union { v16bf v; uint4 u[2]; } B;
      B.u[0] = *(const uint4*)bp;
      B.u[1] = *(const uint4*)(bp + 16);
      acc = WMMA_BF16(A[kk], B.v, acc);
    }

    // D tile: row M = r + mh, col N = n
    float s8 = 0.0f;
#pragma unroll
    for (int r = 0; r < 8; ++r) {
      const int trow = t0 + r + mh;
      const float vv = acc[r];
      s8 += vv;
      if (n0 < 64)        qb [wbase + (size_t)trow * 64 + n]            = (__bf16)vv;
      else if (n0 < 128)  kb [wbase + (size_t)trow * 64 + (n - 64)]     = (__bf16)vv;
      else                vtb[wbase + (size_t)(n - 128) * 1024 + trow]  = (__bf16)vv;
    }
    if (n0 < 64)       atomicAdd(&qsL[n], s8);
    else if (n0 < 128) atomicAdd(&ksL[n - 64], s8);
  }

  __syncthreads();
  if (tid < 64) {
    atomicAdd(&qsumG[win * 64 + tid], qsL[tid]);
    atomicAdd(&ksumG[win * 64 + tid], ksL[tid]);
  }
}

// ---------------- Kernel 2: a_r = relu(q_r k_r^T), 64x64 ----------------
__global__ __launch_bounds__(256)
void BA_ar_kernel(const float* __restrict__ qs, const float* __restrict__ ks,
                  float* __restrict__ ar) {
  __shared__ float qsh[4096], ksh[4096];
  for (int idx = threadIdx.x; idx < 4096; idx += 256) { qsh[idx] = qs[idx]; ksh[idx] = ks[idx]; }
  __syncthreads();
  const float sc = 1.0f / (1024.0f * 1024.0f);   // fold the two means
  for (int p = threadIdx.x; p < 4096; p += 256) {
    const int i = p >> 6, j = p & 63;
    float d = 0.0f;
#pragma unroll
    for (int c = 0; c < 64; ++c) d = fmaf(qsh[i * 64 + c], ksh[j * 64 + c], d);
    d *= sc;
    ar[p] = d > 0.0f ? d : 0.0f;
  }
}

// ---------------- Kernel 3: window mixing q_m/k_m = a_r @ {q,k} ----------------
// grid = (256, 2): x -> 256 tc-chunks, y -> which half of the 64 output windows.
__global__ __launch_bounds__(256)
void BA_mix_kernel(const __bf16* __restrict__ qb, const __bf16* __restrict__ kb,
                   const float* __restrict__ ar,
                   __bf16* __restrict__ qm, __bf16* __restrict__ km) {
  const int tc = blockIdx.x * 256 + threadIdx.x;  // 0..65535 (t*64+c), coalesced
  const int i0 = blockIdx.y * 32;

  for (int pass = 0; pass < 2; ++pass) {
    const __bf16* src = pass ? kb : qb;
    __bf16* dst       = pass ? km : qm;
    float acc[32];
#pragma unroll
    for (int i = 0; i < 32; ++i) acc[i] = 0.0f;
#pragma unroll
    for (int jb = 0; jb < 64; jb += 16) {
      float qv[16];
#pragma unroll
      for (int j = 0; j < 16; ++j) qv[j] = (float)src[(size_t)(jb + j) * 65536 + tc];
#pragma unroll
      for (int i = 0; i < 32; ++i)
#pragma unroll
        for (int j = 0; j < 16; ++j)
          acc[i] = fmaf(ar[(i0 + i) * 64 + jb + j], qv[j], acc[i]);  // uniform -> s_load
    }
#pragma unroll
    for (int i = 0; i < 32; ++i) dst[(size_t)(i0 + i) * 65536 + tc] = (__bf16)acc[i];
  }
}

// ---------------- Kernel 4: fused relu-attention + output (WMMA, async double-buffer) ----
// grid = 64 windows * 8 token-blocks, block = 256 (8 waves), wave = 16 token rows.
__global__ __launch_bounds__(256)
void BA_attn_kernel(const __bf16* __restrict__ qm, const __bf16* __restrict__ km,
                    const __bf16* __restrict__ vtb, float* __restrict__ out) {
  __shared__ __bf16 kmS[2][128 * 72];     // [s][c] tile of k_m, padded stride
  __shared__ __bf16 vtS[2][64 * 136];     // [c][s] tile of v^T, padded stride
  __shared__ __bf16 attnS[8 * 16 * 136];  // per-wave relu(attn) staging [t][s], padded

  const int tid = threadIdx.x, bid = blockIdx.x;
  const int win = bid >> 3, mblk = bid & 7;
  const int wv = tid >> 5, lane = tid & 31, tloc = lane & 15;
  const int t0 = mblk * 128 + wv * 16;
  const size_t wbase = (size_t)win << 16;
  const int khalfA = (lane >> 4) << 3;  // 0 or 8
  const int khB    = (lane >> 4) << 3;  // 0 or 8
  __bf16* attnW = attnS + wv * (16 * 136);

  // issue async staging of one s-tile into buffer `buf`
  auto issue_stage = [&](int sbn, int buf) {
    const __bf16* kmsrc = km + wbase + (size_t)sbn * 8192;   // contiguous [128s][64c]
    for (int q = tid; q < 1024; q += 256) {
      const int s = q >> 3, ch = q & 7;
      async_copy16(kmsrc + q * 8, &kmS[buf][s * 72 + ch * 8]);
    }
    const __bf16* vtsrc = vtb + wbase + sbn * 128;           // 64 rows, stride 1024
    for (int q = tid; q < 1024; q += 256) {
      const int row = q >> 4, col = q & 15;
      async_copy16(vtsrc + (size_t)row * 1024 + col * 8, &vtS[buf][row * 136 + col * 8]);
    }
  };

  // q_m A operands for this wave's 16 rows (held across all s-tiles)
  v16bf Aq[2];
  {
    const __bf16* qp = qm + wbase + (size_t)(t0 + tloc) * 64;
#pragma unroll
    for (int kk = 0; kk < 2; ++kk) {
      const int kb0 = kk * 32 + khalfA;
      union { v16bf v; uint4 u[2]; } tmp;
      tmp.u[0] = *(const uint4*)(qp + kb0);
      tmp.u[1] = *(const uint4*)(qp + kb0 + 16);
      Aq[kk] = tmp.v;
    }
  }

  v8f oacc[4];
#pragma unroll
  for (int ct = 0; ct < 4; ++ct)
#pragma unroll
    for (int e = 0; e < 8; ++e) oacc[ct][e] = 0.0f;

  issue_stage(0, 0);

  for (int sb = 0; sb < 8; ++sb) {
    const int cur = sb & 1;
    wait_async0();        // my async writes for tile sb are in LDS
    __syncthreads();      // everyone's are
    if (sb < 7) issue_stage(sb + 1, cur ^ 1);   // prefetch next tile into other buffer

    // GEMM1: attn tile (16t x 128s) = q_m (16x64) @ k_m^T, relu, stage as A layout
#pragma unroll
    for (int nt = 0; nt < 8; ++nt) {
      const int n = nt * 16 + tloc;                  // s index (B column)
      v8f acc;
#pragma unroll
      for (int e = 0; e < 8; ++e) acc[e] = 0.0f;
#pragma unroll
      for (int kk = 0; kk < 2; ++kk) {
        const __bf16* bp = &kmS[cur][n * 72 + kk * 32 + khB];
        union { v16bf v; uint4 u[2]; } B;
        B.u[0] = *(const uint4*)bp;
        B.u[1] = *(const uint4*)(bp + 16);
        acc = WMMA_BF16(Aq[kk], B.v, acc);
      }
#pragma unroll
      for (int r = 0; r < 8; ++r) {
        float vv = acc[r];
        vv = vv > 0.0f ? vv : 0.0f;
        attnW[(r + khalfA) * 136 + n] = (__bf16)vv;
      }
    }
    __syncthreads();

    // GEMM2: o (16t x 64c) += attn (16x128) @ v (128x64)
#pragma unroll
    for (int kt = 0; kt < 4; ++kt) {
      const int kb0 = kt * 32 + khalfA;
      union { v16bf v; uint4 u[2]; } ta;
      ta.u[0] = *(const uint4*)&attnW[tloc * 136 + kb0];
      ta.u[1] = *(const uint4*)&attnW[tloc * 136 + kb0 + 16];
#pragma unroll
      for (int ct = 0; ct < 4; ++ct) {
        const int c = ct * 16 + tloc;
        const __bf16* vp = &vtS[cur][c * 136 + kt * 32 + khB];
        union { v16bf v; uint4 u[2]; } Bv;
        Bv.u[0] = *(const uint4*)vp;
        Bv.u[1] = *(const uint4*)(vp + 16);
        oacc[ct] = WMMA_BF16(ta.v, Bv.v, oacc[ct]);
      }
    }
  }

  // write output in CHW layout
  const int wh0 = (win >> 3) << 5, ww0 = (win & 7) << 5;
#pragma unroll
  for (int ct = 0; ct < 4; ++ct) {
    const int c = ct * 16 + tloc;
#pragma unroll
    for (int r = 0; r < 8; ++r) {
      const int t = t0 + r + khalfA;
      const int h = wh0 + (t >> 5), w = ww0 + (t & 31);
      out[(size_t)c * 65536 + (size_t)h * 256 + w] = oacc[ct][r];
    }
  }
}

// ---------------- launch ----------------
extern "C" void kernel_launch(void* const* d_in, const int* in_sizes, int n_in,
                              void* d_out, int out_size, void* d_ws, size_t ws_size,
                              hipStream_t stream) {
  const float* x    = (const float*)d_in[0];
  const float* W    = (const float*)d_in[1];
  const float* bias = (const float*)d_in[2];
  float* out = (float*)d_out;

  constexpr size_t SZ = (size_t)64 * 65536;   // elements per (win, 1024, 64) tensor
  __bf16* qb  = (__bf16*)d_ws;
  __bf16* kb  = qb  + SZ;
  __bf16* vtb = kb  + SZ;
  __bf16* qm  = vtb + SZ;
  __bf16* km  = qm  + SZ;
  float* qsum = (float*)(km + SZ);
  float* ksum = qsum + 4096;
  float* ar   = ksum + 4096;

  BA_zero_kernel<<<32, 256, 0, stream>>>(qsum, 8192);          // zeros qsum+ksum
  BA_qkv_kernel<<<512, 256, 0, stream>>>(x, W, bias, qb, kb, vtb, qsum, ksum);
  BA_ar_kernel<<<1, 256, 0, stream>>>(qsum, ksum, ar);
  BA_mix_kernel<<<dim3(256, 2), 256, 0, stream>>>(qb, kb, ar, qm, km);
  BA_attn_kernel<<<512, 256, 0, stream>>>(qm, km, vtb, out);
}
